// PointerNet_59511066853493
// MI455X (gfx1250) — compile-verified, matrix operand order checked by reading, not run on hardware
//
#include <hip/hip_runtime.h>
#include <math.h>

#define B_ 256
#define S_ 128
#define E_ 128
#define H_ 512
#define G_ 2048   /* 4*H */
#define NEGV (-10000.0f)

typedef __bf16 bf16;
typedef __attribute__((ext_vector_type(16))) __bf16 v16bf;
typedef __attribute__((ext_vector_type(8)))  __bf16 v8bf;
typedef __attribute__((ext_vector_type(8)))  float  v8f;

// ---------------------------------------------------------------------------
// Fragment loader for V_WMMA_F32_16X16X32_BF16.
// 16-bit A-matrix 16x32 layout (ISA 7.12.2): lanes 0-15 hold row M=lane with
// K = {0..7, 16..23}; lanes 16-31 hold the same rows with K = {8..15, 24..31}.
// B (32x16) is loaded with the mirrored pattern from row-major W[N,K] since
// B[k,n] = W[n,k]: lane's "row" is the W row n.
// ---------------------------------------------------------------------------
__device__ __forceinline__ v16bf load_frag(const bf16* __restrict__ base, long ld,
                                           int row, int k0, int half) {
  const bf16* p = base + (long)row * ld + k0 + half * 8;
  v8bf lo = *(const v8bf*)(p);
  v8bf hi = *(const v8bf*)(p + 16);
  v16bf r;
#pragma unroll
  for (int i = 0; i < 8; ++i) { r[i] = lo[i]; r[i + 8] = hi[i]; }
  return r;
}

// C[M,N] = A1[M,K1] @ Wa[N,K1]^T + A2[M,K2] @ Wb[N,K2]^T + bias[N]   (f32 out)
// One 16x32 C tile per wave (A fragment reused across two WMMAs).
// K2==0 skips the second pair (q-projection use). Ntiles = N/32.
__global__ void k_gemm_dual(const bf16* __restrict__ A1, long ldA1,
                            const bf16* __restrict__ Wa, int K1,
                            const bf16* __restrict__ A2, long ldA2,
                            const bf16* __restrict__ Wb, int K2,
                            const float* __restrict__ bias,
                            float* __restrict__ C, long ldC, int Ntiles) {
  const int wave = threadIdx.x >> 5;
  const int lane = threadIdx.x & 31;
  const int laneLo = lane & 15;
  const int half = lane >> 4;
  const int tile = blockIdx.x * (blockDim.x >> 5) + wave;
  const int nt = tile % Ntiles, mt = tile / Ntiles;
  const int m0 = mt * 16, n0 = nt * 32;

  v8f acc0 = {0.f, 0.f, 0.f, 0.f, 0.f, 0.f, 0.f, 0.f};
  v8f acc1 = {0.f, 0.f, 0.f, 0.f, 0.f, 0.f, 0.f, 0.f};
  for (int k0 = 0; k0 < K1; k0 += 32) {
    v16bf a  = load_frag(A1, ldA1, m0 + laneLo, k0, half);
    v16bf b0 = load_frag(Wa, K1, n0 + laneLo, k0, half);
    v16bf b1 = load_frag(Wa, K1, n0 + 16 + laneLo, k0, half);
    acc0 = __builtin_amdgcn_wmma_f32_16x16x32_bf16(false, a, false, b0,
                                                   (short)0, acc0, false, false);
    acc1 = __builtin_amdgcn_wmma_f32_16x16x32_bf16(false, a, false, b1,
                                                   (short)0, acc1, false, false);
  }
  for (int k0 = 0; k0 < K2; k0 += 32) {
    v16bf a  = load_frag(A2, ldA2, m0 + laneLo, k0, half);
    v16bf b0 = load_frag(Wb, K2, n0 + laneLo, k0, half);
    v16bf b1 = load_frag(Wb, K2, n0 + 16 + laneLo, k0, half);
    acc0 = __builtin_amdgcn_wmma_f32_16x16x32_bf16(false, a, false, b0,
                                                   (short)0, acc0, false, false);
    acc1 = __builtin_amdgcn_wmma_f32_16x16x32_bf16(false, a, false, b1,
                                                   (short)0, acc1, false, false);
  }
  const float bc0 = bias[n0 + laneLo];
  const float bc1 = bias[n0 + 16 + laneLo];
#pragma unroll
  for (int j = 0; j < 8; ++j) {  // C layout: VGPR j -> M = j (+8 for lane half 1)
    float* row = C + (long)(m0 + j + half * 8) * ldC;
    row[n0 + laneLo] = acc0[j] + bc0;
    row[n0 + 16 + laneLo] = acc1[j] + bc1;
  }
}

// Same GEMM (single pair) with bf16 output, strided into [B,S,H]. Ntiles=N/32.
__global__ void k_gemm_bf16out(const bf16* __restrict__ A, long ldA,
                               const bf16* __restrict__ W, int K,
                               const float* __restrict__ bias,
                               bf16* __restrict__ C, long ldC, int Ntiles) {
  const int wave = threadIdx.x >> 5;
  const int lane = threadIdx.x & 31;
  const int laneLo = lane & 15;
  const int half = lane >> 4;
  const int tile = blockIdx.x * (blockDim.x >> 5) + wave;
  const int nt = tile % Ntiles, mt = tile / Ntiles;
  const int m0 = mt * 16, n0 = nt * 32;

  v8f acc0 = {0.f, 0.f, 0.f, 0.f, 0.f, 0.f, 0.f, 0.f};
  v8f acc1 = {0.f, 0.f, 0.f, 0.f, 0.f, 0.f, 0.f, 0.f};
  for (int k0 = 0; k0 < K; k0 += 32) {
    v16bf a  = load_frag(A, ldA, m0 + laneLo, k0, half);
    v16bf b0 = load_frag(W, K, n0 + laneLo, k0, half);
    v16bf b1 = load_frag(W, K, n0 + 16 + laneLo, k0, half);
    acc0 = __builtin_amdgcn_wmma_f32_16x16x32_bf16(false, a, false, b0,
                                                   (short)0, acc0, false, false);
    acc1 = __builtin_amdgcn_wmma_f32_16x16x32_bf16(false, a, false, b1,
                                                   (short)0, acc1, false, false);
  }
  const float bc0 = bias[n0 + laneLo];
  const float bc1 = bias[n0 + 16 + laneLo];
#pragma unroll
  for (int j = 0; j < 8; ++j) {
    bf16* row = C + (long)(m0 + j + half * 8) * ldC;
    row[n0 + laneLo] = (bf16)(acc0[j] + bc0);
    row[n0 + 16 + laneLo] = (bf16)(acc1[j] + bc1);
  }
}

// Pointwise LSTM cell (gate order i,f,g,o). Encoder variant (seqlen != null)
// freezes state past sequence length and emits houtBf = valid ? h2 : 0.
__global__ void k_lstm_cell(const float* __restrict__ gates,
                            float* __restrict__ h, float* __restrict__ c,
                            bf16* __restrict__ hBf, bf16* __restrict__ houtBf,
                            const int* __restrict__ seqlen, int t) {
  const int gid = blockIdx.x * blockDim.x + threadIdx.x;  // B*H
  const int b = gid >> 9;
  const float* g = gates + (long)b * G_ + (gid & (H_ - 1));
  const float iv = g[0], fv = g[H_], gv = g[2 * H_], ov = g[3 * H_];
  const float is = 1.f / (1.f + expf(-iv));
  const float fs = 1.f / (1.f + expf(-fv));
  const float os = 1.f / (1.f + expf(-ov));
  const float c2 = fs * c[gid] + is * tanhf(gv);
  const float h2 = os * tanhf(c2);
  if (seqlen) {
    const bool valid = t < seqlen[b];
    const float hn = valid ? h2 : h[gid];
    const float cn = valid ? c2 : c[gid];
    h[gid] = hn; c[gid] = cn; hBf[gid] = (bf16)hn;
    houtBf[gid] = (bf16)(valid ? h2 : 0.f);
  } else {
    h[gid] = h2; c[gid] = c2; hBf[gid] = (bf16)h2;
  }
}

// Attention + masked softmax + first-index argmax + next-input gather.
// One block (256 threads = 8 waves) per batch row. Each lane reads a
// contiguous 32B (16 bf16) chunk of the encProj row -> coalesced b128 loads;
// next s-row is prefetched (global_prefetch_b8) to keep the HBM stream ahead.
__global__ void k_attention(const bf16* __restrict__ encProj,  // [B,S,H]
                            const float* __restrict__ q,       // [B,H]
                            const float* __restrict__ Vv,      // [H]
                            int* __restrict__ mask,            // [B,S]
                            const bf16* __restrict__ embBf,    // [B,S,E]
                            bf16* __restrict__ xBf,            // [B,E]
                            float* __restrict__ probsT,        // d_out + t*B*S
                            float* __restrict__ idxOut,        // d_out + S*B*S
                            int t) {
  __shared__ float qs[H_];
  __shared__ float vsh[H_];
  __shared__ float vs[S_];
  __shared__ float es[S_];
  __shared__ float redM, redS;
  __shared__ int bestSh;
  const int b = blockIdx.x;
  const int tid = threadIdx.x;
  const int lane = tid & 31, wave = tid >> 5;

  qs[tid] = q[b * H_ + tid];
  qs[tid + 256] = q[b * H_ + tid + 256];
  vsh[tid] = Vv[tid];
  vsh[tid + 256] = Vv[tid + 256];
  __syncthreads();

  const bf16* ep = encProj + (long)b * S_ * H_;
  const int j0 = lane * 16;                     // 16 contiguous bf16 per lane
  for (int s = wave; s < S_; s += 8) {
    const bf16* p = ep + (long)s * H_ + j0;
    if (s + 8 < S_)
      __builtin_prefetch(ep + (long)(s + 8) * H_ + j0, 0, 3);
    v8bf a0 = *(const v8bf*)(p);
    v8bf a1 = *(const v8bf*)(p + 8);
    float sum = 0.f;
#pragma unroll
    for (int i = 0; i < 8; ++i) {
      sum += vsh[j0 + i] * tanhf((float)a0[i] + qs[j0 + i]);
      sum += vsh[j0 + 8 + i] * tanhf((float)a1[i] + qs[j0 + 8 + i]);
    }
    for (int off = 16; off; off >>= 1) sum += __shfl_xor(sum, off, 32);
    if (lane == 0) vs[s] = mask[b * S_ + s] ? sum : NEGV;
  }
  __syncthreads();

  if (wave == 0) {                       // max over S=128 (4 per lane)
    float m = -INFINITY;
#pragma unroll
    for (int k = 0; k < 4; ++k) m = fmaxf(m, vs[lane + 32 * k]);
    for (int off = 16; off; off >>= 1) m = fmaxf(m, __shfl_xor(m, off, 32));
    if (lane == 0) redM = m;
  }
  __syncthreads();
  if (tid < S_) es[tid] = expf(vs[tid] - redM);
  __syncthreads();
  if (wave == 0) {                       // sum + argmax (first max on ties)
    float sum = 0.f;
    float bv = -INFINITY; int bi = 0;
#pragma unroll
    for (int k = 0; k < 4; ++k) {
      const int s = lane + 32 * k;
      sum += es[s];
      const float v = vs[s];
      if (v > bv) { bv = v; bi = s; }
    }
    for (int off = 16; off; off >>= 1) {
      sum += __shfl_xor(sum, off, 32);
      const float ov = __shfl_xor(bv, off, 32);
      const int   oi = __shfl_xor(bi, off, 32);
      if (ov > bv || (ov == bv && oi < bi)) { bv = ov; bi = oi; }
    }
    if (lane == 0) { redS = sum; bestSh = bi; }
  }
  __syncthreads();

  if (tid < S_) probsT[b * S_ + tid] = es[tid] / redS;
  const int best = bestSh;
  if (tid == 0) {
    mask[b * S_ + best] = 0;
    idxOut[b * S_ + t] = (float)best;
  }
  if (tid < E_) xBf[b * E_ + tid] = embBf[((long)b * S_ + best) * E_ + tid];
}

__global__ void k_embed(const float* __restrict__ in, const float* __restrict__ Wemb,
                        const float* __restrict__ bemb, bf16* __restrict__ embBf) {
  const int gid = blockIdx.x * blockDim.x + threadIdx.x;  // B*S*E
  const int e = gid & (E_ - 1);
  const int bs = gid >> 7;
  const float x0 = in[bs * 2], x1 = in[bs * 2 + 1];
  embBf[gid] = (bf16)(x0 * Wemb[2 * e] + x1 * Wemb[2 * e + 1] + bemb[e]);
}

__global__ void k_init(const float* __restrict__ h0, const float* __restrict__ c0,
                       const float* __restrict__ dec0, const float* __restrict__ in,
                       float* __restrict__ h, float* __restrict__ c,
                       bf16* __restrict__ hBf, bf16* __restrict__ xBf,
                       int* __restrict__ mask) {
  const int gid = blockIdx.x * blockDim.x + threadIdx.x;  // B*H
  const int j = gid & (H_ - 1);
  h[gid] = h0[j]; c[gid] = c0[j]; hBf[gid] = (bf16)h0[j];
  if (gid < B_ * E_) xBf[gid] = (bf16)dec0[gid & (E_ - 1)];
  if (gid < B_ * S_) mask[gid] = (in[gid * 2] > 0.f) ? 1 : 0;
}

__global__ void k_cvt(const float* __restrict__ s, bf16* __restrict__ d, int n) {
  const int i = blockIdx.x * blockDim.x + threadIdx.x;
  if (i < n) d[i] = (bf16)s[i];
}

__global__ void k_bias2(const float* __restrict__ a, const float* __restrict__ b,
                        float* __restrict__ o, int n) {
  const int i = blockIdx.x * blockDim.x + threadIdx.x;
  if (i < n) o[i] = a[i] + b[i];
}

extern "C" void kernel_launch(void* const* d_in, const int* in_sizes, int n_in,
                              void* d_out, int out_size, void* d_ws, size_t ws_size,
                              hipStream_t stream) {
  (void)in_sizes; (void)n_in; (void)out_size; (void)ws_size;
  const float* inputs  = (const float*)d_in[0];
  const int*   seqlen  = (const int*)d_in[1];
  /* d_in[2] = max_sequence_length (scalar, S_=128 compile-time) */
  const float* W_emb   = (const float*)d_in[3];
  const float* b_emb   = (const float*)d_in[4];
  const float* enc_Wih = (const float*)d_in[5];
  const float* enc_Whh = (const float*)d_in[6];
  const float* enc_bih = (const float*)d_in[7];
  const float* enc_bhh = (const float*)d_in[8];
  const float* dec_Wih = (const float*)d_in[9];
  const float* dec_Whh = (const float*)d_in[10];
  const float* dec_bih = (const float*)d_in[11];
  const float* dec_bhh = (const float*)d_in[12];
  const float* h0      = (const float*)d_in[13];
  const float* c0      = (const float*)d_in[14];
  const float* dec0    = (const float*)d_in[15];
  const float* W1w     = (const float*)d_in[16];
  const float* b1      = (const float*)d_in[17];
  const float* W2w     = (const float*)d_in[18];
  const float* b2      = (const float*)d_in[19];
  const float* Vv      = (const float*)d_in[20];

  char* ws = (char*)d_ws;
  size_t off = 0;
  auto alloc = [&](size_t bytes) -> char* {
    char* p = ws + off;
    off += (bytes + 255) & ~(size_t)255;
    return p;
  };
  bf16* embBf     = (bf16*)alloc((size_t)B_ * S_ * E_ * 2);   // 8.4 MB
  bf16* encProjBf = (bf16*)alloc((size_t)B_ * S_ * H_ * 2);   // 33.5 MB
  bf16* wEncIh    = (bf16*)alloc((size_t)G_ * E_ * 2);
  bf16* wEncHh    = (bf16*)alloc((size_t)G_ * H_ * 2);
  bf16* wDecIh    = (bf16*)alloc((size_t)G_ * E_ * 2);
  bf16* wDecHh    = (bf16*)alloc((size_t)G_ * H_ * 2);
  bf16* w1b       = (bf16*)alloc((size_t)H_ * H_ * 2);
  bf16* w2b       = (bf16*)alloc((size_t)H_ * H_ * 2);
  float* encB     = (float*)alloc((size_t)G_ * 4);
  float* decB     = (float*)alloc((size_t)G_ * 4);
  float* gates    = (float*)alloc((size_t)B_ * G_ * 4);       // 2 MB
  float* hBuf     = (float*)alloc((size_t)B_ * H_ * 4);
  float* cBuf     = (float*)alloc((size_t)B_ * H_ * 4);
  bf16* hBf       = (bf16*)alloc((size_t)B_ * H_ * 2);
  bf16* houtBf    = (bf16*)alloc((size_t)B_ * H_ * 2);
  float* qBuf     = (float*)alloc((size_t)B_ * H_ * 4);
  bf16* xBf       = (bf16*)alloc((size_t)B_ * E_ * 2);
  int*  mask      = (int*)alloc((size_t)B_ * S_ * 4);

  float* probs  = (float*)d_out;                         // [S,B,S]
  float* idxOut = probs + (size_t)S_ * B_ * S_;          // [B,S] as float

  // --- per-call prep (re-done every call: deterministic graph replay) ---
  k_cvt<<<(G_ * E_ + 255) / 256, 256, 0, stream>>>(enc_Wih, wEncIh, G_ * E_);
  k_cvt<<<(G_ * H_ + 255) / 256, 256, 0, stream>>>(enc_Whh, wEncHh, G_ * H_);
  k_cvt<<<(G_ * E_ + 255) / 256, 256, 0, stream>>>(dec_Wih, wDecIh, G_ * E_);
  k_cvt<<<(G_ * H_ + 255) / 256, 256, 0, stream>>>(dec_Whh, wDecHh, G_ * H_);
  k_cvt<<<(H_ * H_ + 255) / 256, 256, 0, stream>>>(W1w, w1b, H_ * H_);
  k_cvt<<<(H_ * H_ + 255) / 256, 256, 0, stream>>>(W2w, w2b, H_ * H_);
  k_bias2<<<(G_ + 255) / 256, 256, 0, stream>>>(enc_bih, enc_bhh, encB, G_);
  k_bias2<<<(G_ + 255) / 256, 256, 0, stream>>>(dec_bih, dec_bhh, decB, G_);
  k_embed<<<(B_ * S_ * E_) / 256, 256, 0, stream>>>(inputs, W_emb, b_emb, embBf);
  k_init<<<(B_ * H_) / 256, 256, 0, stream>>>(h0, c0, dec0, inputs,
                                              hBuf, cBuf, hBf, xBf, mask);

  // Gates GEMM: M=256,N=2048 -> (16 x 64)=1024 wave-tiles, 4 waves/block -> 256 blocks
  // Proj/Q GEMM: M=256,N=512 -> (16 x 16)=256 wave-tiles -> 64 blocks
  const int gatesBlocks = 256, gatesNt = G_ / 32;
  const int smallBlocks = 64, smallNt = H_ / 32;

  // ----------------------------- encoder -----------------------------------
  for (int t = 0; t < S_; ++t) {
    k_gemm_dual<<<gatesBlocks, 128, 0, stream>>>(
        embBf + (long)t * E_, (long)S_ * E_, wEncIh, E_,
        hBf, (long)H_, wEncHh, H_,
        encB, gates, (long)G_, gatesNt);
    k_lstm_cell<<<(B_ * H_) / 256, 256, 0, stream>>>(
        gates, hBuf, cBuf, hBf, houtBf, seqlen, t);
    k_gemm_bf16out<<<smallBlocks, 128, 0, stream>>>(
        houtBf, (long)H_, w1b, H_, b1,
        encProjBf + (long)t * H_, (long)S_ * H_, smallNt);
  }

  // ----------------------------- decoder -----------------------------------
  for (int t = 0; t < S_; ++t) {
    k_gemm_dual<<<gatesBlocks, 128, 0, stream>>>(
        xBf, (long)E_, wDecIh, E_,
        hBf, (long)H_, wDecHh, H_,
        decB, gates, (long)G_, gatesNt);
    k_lstm_cell<<<(B_ * H_) / 256, 256, 0, stream>>>(
        gates, hBuf, cBuf, hBf, nullptr, nullptr, t);
    k_gemm_dual<<<smallBlocks, 128, 0, stream>>>(
        hBf, (long)H_, w2b, H_,
        nullptr, 0, nullptr, 0,
        b2, qBuf, (long)H_, smallNt);
    k_attention<<<B_, 256, 0, stream>>>(
        encProjBf, qBuf, Vv, mask, embBf, xBf,
        probs + (size_t)t * B_ * S_, idxOut, t);
  }
}
